// ImportanceRenderer_72808285601794
// MI455X (gfx1250) — matrix operand order repaired and checked by reading, not compiled, not run on hardware
//
#include <hip/hip_runtime.h>
#include <hip/hip_bf16.h>
#include <math.h>

// Problem constants (from reference)
#define N_B      2
#define M_RAYS   4096
#define D_SAMP   48
#define C_CH     32
#define HID_N    64
#define RES      256
#define NRAYS    (N_B * M_RAYS)          // 8192
#define NSAMP    (NRAYS * D_SAMP)        // 393216 samples per pass
#define DELTA_Z  ((2.5f - 0.5f) / (float)(D_SAMP - 1))

typedef __attribute__((ext_vector_type(16))) _Float16 v16h;
typedef __attribute__((ext_vector_type(8)))  float    v8f;

// Branchless softplus using hardware v_exp_f32 / v_log_f32:
// softplus(x) = max(x,0) + log(1 + exp(-|x|)).  Stable for all x, no EXEC
// divergence, and the TRANS ops co-execute with the XDL WMMA pipe.
__device__ __forceinline__ float softplus_f(float x) {
  return fmaxf(x, 0.f) + __logf(1.f + __expf(-fabsf(x)));
}
__device__ __forceinline__ float sigmoid_f(float x) {
  return 1.f / (1.f + __expf(-x));
}
// 16-bit A-matrix 16x32 fragment: which K does element i of this lane hold
__device__ __forceinline__ int chanK(int i, int hi) {
  return hi ? (i < 8 ? i + 8 : i + 16) : (i < 8 ? i : i + 8);
}

// ---------------------------------------------------------------------------
// Kernel 0: stratified coarse depths  z = 0.5 + d*delta + noise*delta
// ---------------------------------------------------------------------------
__global__ void coarse_depths_kernel(const float* __restrict__ noise,
                                     float* __restrict__ depths) {
  int i = blockIdx.x * blockDim.x + threadIdx.x;
  if (i >= NSAMP) return;
  int d = i % D_SAMP;
  depths[i] = 0.5f + (float)d * DELTA_Z + noise[i] * DELTA_Z;
}

// ---------------------------------------------------------------------------
// Kernel 1/3: fused tri-plane gather + WMMA MLP decode.
// One wave processes 16 samples. Lanes l and l+16 co-own sample (l&15);
// each lane gathers exactly the 16 channels its WMMA A-fragment holds.
// GEMM1: (16x32 f16) @ (32x64) via 4x v_wmma_f32_16x16x32_f16.
// GEMM2: hidden re-fragmented through LDS, W2 zero-padded to 64x16,
//        2x v_wmma_f32_16x16x32_f16 accumulating.
// ---------------------------------------------------------------------------
__global__ __launch_bounds__(256) void render_pass_kernel(
    const float* __restrict__ planes,   // [N,3,32,256,256]
    const float* __restrict__ ray_o,    // [N,M,3]
    const float* __restrict__ ray_d,    // [N,M,3]
    const float* __restrict__ depths,   // [NSAMP]
    const float* __restrict__ W1,       // [32,64]
    const float* __restrict__ b1,       // [64]
    const float* __restrict__ W2,       // [64,4]
    const float* __restrict__ b2,       // [4]
    float* __restrict__ rgb_out,        // [NSAMP,3]
    float* __restrict__ sig_out)        // [NSAMP]
{
  __shared__ _Float16 lds_h[8][16][HID_N + 8]; // hidden re-fragment buffer
  __shared__ float    lds_d2[8][16];           // |coords|^2 per sample

  const int lane  = threadIdx.x & 31;
  const int wIB   = threadIdx.x >> 5;
  const int hi    = lane >> 4;        // 0: lanes 0-15, 1: lanes 16-31
  const int ln    = lane & 15;
  const int khalf = hi ? 16 : 0;

  __builtin_prefetch(W1, 0, 3);
  __builtin_prefetch(W2, 0, 3);

  // --- Stage MLP weights into WMMA B fragments (f16), once per wave ------
  // B 32x16 f16: lanes 0-15 hold K=0..7,16..23 (col = ln), lanes 16-31 K=8..15,24..31.
  v16h bw1[4];
  #pragma unroll
  for (int t = 0; t < 4; ++t)
    #pragma unroll
    for (int i = 0; i < 16; ++i)
      bw1[t][i] = (_Float16)W1[chanK(i, hi) * HID_N + 16 * t + ln];

  v16h bw2[2]; // W2 (64x4) zero-padded to 64x16, split K into 2 chunks of 32
  #pragma unroll
  for (int kc = 0; kc < 2; ++kc)
    #pragma unroll
    for (int i = 0; i < 16; ++i)
      bw2[kc][i] = (ln < 4) ? (_Float16)W2[(kc * 32 + chanK(i, hi)) * 4 + ln]
                            : (_Float16)0.f;

  float b1v[4];
  #pragma unroll
  for (int t = 0; t < 4; ++t) b1v[t] = b1[16 * t + ln];
  const float b2v = (ln < 4) ? b2[ln] : 0.f;

  const int totalWaves = (int)((gridDim.x * blockDim.x) >> 5);
  const int wave0      = (int)((blockIdx.x * blockDim.x + threadIdx.x) >> 5);
  const int numGroups  = NSAMP / 16;

  for (int g = wave0; g < numGroups; g += totalWaves) {
    const int gs = g * 16 + ln;          // this lane's sample id
    const int rg = gs / D_SAMP;          // global ray id
    const int n  = rg / M_RAYS;          // batch index

    const float dep = depths[gs];
    const float ox = ray_o[rg * 3 + 0], oy = ray_o[rg * 3 + 1], oz = ray_o[rg * 3 + 2];
    const float dx = ray_d[rg * 3 + 0], dy = ray_d[rg * 3 + 1], dz = ray_d[rg * 3 + 2];
    const float px = ox + dep * dx, py = oy + dep * dy, pz = oz + dep * dz;
    if (!hi) lds_d2[wIB][ln] = px * px + py * py + pz * pz;

    // --- tri-plane bilinear gather: this lane's 16 channels -------------
    float acc[16];
    #pragma unroll
    for (int i = 0; i < 16; ++i) acc[i] = 0.f;

    #pragma unroll
    for (int p = 0; p < 3; ++p) {
      // plane projections: p0:(x,y) p1:(x,z) p2:(z,y)
      const float gx = (p == 2) ? pz : px;
      const float gy = (p == 1) ? pz : py;
      const float fx = ((gx + 1.f) * (float)RES - 1.f) * 0.5f;
      const float fy = ((gy + 1.f) * (float)RES - 1.f) * 0.5f;
      const float x0f = floorf(fx), y0f = floorf(fy);
      const float wx = fx - x0f, wy = fy - y0f;
      const int x0 = (int)x0f, y0 = (int)y0f;
      const float* pb = planes + (size_t)(n * 3 + p) * C_CH * RES * RES;
      #pragma unroll
      for (int tp = 0; tp < 4; ++tp) {
        const int xi = x0 + (tp & 1);
        const int yi = y0 + (tp >> 1);
        float wt = ((tp & 1) ? wx : 1.f - wx) * ((tp >> 1) ? wy : 1.f - wy);
        const bool ok = (xi >= 0) & (xi < RES) & (yi >= 0) & (yi < RES);
        wt = ok ? wt : 0.f;
        const int xc = min(max(xi, 0), RES - 1);
        const int yc = min(max(yi, 0), RES - 1);
        const float* tap = pb + yc * RES + xc;
        __builtin_prefetch(tap, 0, 0);
        #pragma unroll
        for (int i = 0; i < 16; ++i) {
          const int c = chanK(i, hi);
          acc[i] += wt * tap[(size_t)c * RES * RES];
        }
      }
    }

    // features = mean over 3 planes -> f16 A fragment
    v16h a;
    #pragma unroll
    for (int i = 0; i < 16; ++i) a[i] = (_Float16)(acc[i] * (1.f / 3.f));

    // --- GEMM1: X(16x32) @ W1(32x64), 4 WMMA tiles ----------------------
    v8f hacc[4];
    #pragma unroll
    for (int t = 0; t < 4; ++t) {
      v8f cz = {};
      hacc[t] = __builtin_amdgcn_wmma_f32_16x16x32_f16(
          false, a, false, bw1[t], (short)0, cz, false, false);
    }

    // bias + softplus; re-fragment hidden through LDS (C/D layout -> A layout)
    #pragma unroll
    for (int t = 0; t < 4; ++t)
      #pragma unroll
      for (int r = 0; r < 8; ++r) {
        const float v = softplus_f(hacc[t][r] + b1v[t]);
        lds_h[wIB][r + 8 * hi][16 * t + ln] = (_Float16)v;
      }
    asm volatile("s_wait_dscnt 0" ::: "memory");
    __builtin_amdgcn_wave_barrier();

    v16h a2[2];
    #pragma unroll
    for (int kc = 0; kc < 2; ++kc)
      #pragma unroll
      for (int i = 0; i < 16; ++i)
        a2[kc][i] = lds_h[wIB][ln][kc * 32 + chanK(i, hi)];

    // --- GEMM2: H(16x64) @ W2pad(64x16), 2 accumulating WMMAs -----------
    v8f o = {};
    o = __builtin_amdgcn_wmma_f32_16x16x32_f16(false, a2[0], false, bw2[0],
                                               (short)0, o, false, false);
    o = __builtin_amdgcn_wmma_f32_16x16x32_f16(false, a2[1], false, bw2[1],
                                               (short)0, o, false, false);

    // --- activations + writeback (cols 0..3 valid) ----------------------
    if (ln < 4) {
      #pragma unroll
      for (int r = 0; r < 8; ++r) {
        const int ms  = r + 8 * hi;
        const int sid = g * 16 + ms;
        const float val = o[r] + b2v;
        if (ln == 0) {
          const float d2   = lds_d2[wIB][ms];
          const float blob = 10.f - 20.f * sqrtf(d2); // 10*(1 - sqrt(d2)/0.5)
          sig_out[sid] = val + blob;
        } else {
          rgb_out[sid * 3 + (ln - 1)] = sigmoid_f(val) * 1.002f - 0.001f;
        }
      }
    }
    __builtin_amdgcn_wave_barrier();
  }
}

// ---------------------------------------------------------------------------
// Kernel 2: per-ray coarse ray-march -> smoothed PDF -> inverse-CDF sampling
// ---------------------------------------------------------------------------
__global__ void importance_kernel(const float* __restrict__ depths_c,
                                  const float* __restrict__ sig_c,
                                  const float* __restrict__ u_imp,
                                  float* __restrict__ depths_f) {
  const int r = blockIdx.x * blockDim.x + threadIdx.x;
  if (r >= NRAYS) return;

  float z[D_SAMP], s[D_SAMP];
  for (int i = 0; i < D_SAMP; ++i) {
    z[i] = depths_c[r * D_SAMP + i];
    s[i] = sig_c[r * D_SAMP + i];
  }
  // coarse weights (47)
  float w[D_SAMP - 1];
  float T = 1.f;
  for (int i = 0; i < D_SAMP - 1; ++i) {
    const float dm    = softplus_f(0.5f * (s[i] + s[i + 1]) - 1.f);
    const float alpha = 1.f - __expf(-dm * (z[i + 1] - z[i]));
    w[i] = alpha * T;
    T *= (1.f - alpha + 1e-10f);
  }
  // max-pool smoothing (pad -inf)
  const float NEG_INF = -__builtin_huge_valf();
  float wmax[D_SAMP];
  for (int i = 0; i < D_SAMP; ++i) {
    const float a = (i == 0) ? NEG_INF : w[i - 1];
    const float b = (i == D_SAMP - 1) ? NEG_INF : w[i];
    wmax[i] = fmaxf(a, b);
  }
  float w2[D_SAMP - 1], zm[D_SAMP - 1];
  for (int i = 0; i < D_SAMP - 1; ++i) {
    w2[i] = 0.5f * (wmax[i] + wmax[i + 1]) + 0.01f;
    zm[i] = 0.5f * (z[i] + z[i + 1]);
  }
  // CDF over w2[1..45] (45 weights), cdf length 46
  float cdf[46];
  cdf[0] = 0.f;
  float tot = 0.f, wt[45];
  for (int j = 0; j < 45; ++j) { wt[j] = w2[j + 1] + 1e-5f; tot += wt[j]; }
  const float inv = 1.f / tot;
  for (int j = 0; j < 45; ++j) cdf[j + 1] = cdf[j] + wt[j] * inv;

  for (int k = 0; k < D_SAMP; ++k) {
    const float u = u_imp[r * D_SAMP + k];
    int ind = 0;
    for (int j = 0; j < 46; ++j) ind += (cdf[j] <= u) ? 1 : 0; // searchsorted right
    int below = ind - 1; if (below < 0) below = 0;
    const int above = (ind < 45) ? ind : 45;
    const float cb = cdf[below], ca = cdf[above];
    const float bb = zm[below],  ba = zm[above];
    float den = ca - cb;
    if (den < 1e-5f) den = 1.f;
    depths_f[r * D_SAMP + k] = bb + (u - cb) / den * (ba - bb);
  }
}

// ---------------------------------------------------------------------------
// Kernel 4: merge coarse+fine (sort 96 by depth), final volumetric composite
// ---------------------------------------------------------------------------
__global__ void composite_kernel(const float* __restrict__ depths_c,
                                 const float* __restrict__ rgb_c,
                                 const float* __restrict__ sig_c,
                                 const float* __restrict__ depths_f,
                                 const float* __restrict__ rgb_f,
                                 const float* __restrict__ sig_f,
                                 float* __restrict__ out) {
  const int r = blockIdx.x * blockDim.x + threadIdx.x;
  if (r >= NRAYS) return;
  const int D2 = 2 * D_SAMP;

  float z[2 * D_SAMP], s[2 * D_SAMP];
  float c0[2 * D_SAMP], c1[2 * D_SAMP], c2[2 * D_SAMP];
  for (int i = 0; i < D_SAMP; ++i) {
    const int gi = r * D_SAMP + i;
    z[i] = depths_c[gi]; s[i] = sig_c[gi];
    c0[i] = rgb_c[gi * 3 + 0]; c1[i] = rgb_c[gi * 3 + 1]; c2[i] = rgb_c[gi * 3 + 2];
    z[i + D_SAMP] = depths_f[gi]; s[i + D_SAMP] = sig_f[gi];
    c0[i + D_SAMP] = rgb_f[gi * 3 + 0];
    c1[i + D_SAMP] = rgb_f[gi * 3 + 1];
    c2[i + D_SAMP] = rgb_f[gi * 3 + 2];
  }
  // insertion sort by depth
  for (int i = 1; i < D2; ++i) {
    const float zk = z[i], sk = s[i], a0 = c0[i], a1 = c1[i], a2v = c2[i];
    int j = i - 1;
    while (j >= 0 && z[j] > zk) {
      z[j + 1] = z[j]; s[j + 1] = s[j];
      c0[j + 1] = c0[j]; c1[j + 1] = c1[j]; c2[j + 1] = c2[j];
      --j;
    }
    z[j + 1] = zk; s[j + 1] = sk; c0[j + 1] = a0; c1[j + 1] = a1; c2[j + 1] = a2v;
  }
  // ray march
  float T = 1.f, R = 0.f, G = 0.f, B = 0.f, wsum = 0.f, wz = 0.f;
  for (int i = 0; i < D2 - 1; ++i) {
    const float dlt   = z[i + 1] - z[i];
    const float dm    = softplus_f(0.5f * (s[i] + s[i + 1]) - 1.f);
    const float alpha = 1.f - __expf(-dm * dlt);
    const float wgt   = alpha * T;
    T *= (1.f - alpha + 1e-10f);
    R += wgt * 0.5f * (c0[i] + c0[i + 1]);
    G += wgt * 0.5f * (c1[i] + c1[i + 1]);
    B += wgt * 0.5f * (c2[i] + c2[i + 1]);
    wsum += wgt;
    wz += wgt * 0.5f * (z[i] + z[i + 1]);
  }
  float depth = wz / wsum;
  if (!(depth == depth)) depth = __builtin_huge_valf();
  depth = fminf(fmaxf(depth, z[0]), z[D2 - 1]);

  out[r * 5 + 0] = R * 2.f - 1.f;
  out[r * 5 + 1] = G * 2.f - 1.f;
  out[r * 5 + 2] = B * 2.f - 1.f;
  out[r * 5 + 3] = depth;
  out[r * 5 + 4] = wsum;
}

// ---------------------------------------------------------------------------
extern "C" void kernel_launch(void* const* d_in, const int* in_sizes, int n_in,
                              void* d_out, int out_size, void* d_ws, size_t ws_size,
                              hipStream_t stream) {
  const float* planes = (const float*)d_in[0];
  const float* ray_o  = (const float*)d_in[1];
  const float* ray_d  = (const float*)d_in[2];
  const float* noise  = (const float*)d_in[3];
  const float* u_imp  = (const float*)d_in[4];
  const float* W1     = (const float*)d_in[5];
  const float* b1     = (const float*)d_in[6];
  const float* W2     = (const float*)d_in[7];
  const float* b2     = (const float*)d_in[8];
  float* out = (float*)d_out;

  float* ws    = (float*)d_ws;            // needs 10*NSAMP floats (~15.7 MB)
  float* dep_c = ws;               ws += NSAMP;
  float* rgb_c = ws;               ws += NSAMP * 3;
  float* sig_c = ws;               ws += NSAMP;
  float* dep_f = ws;               ws += NSAMP;
  float* rgb_f = ws;               ws += NSAMP * 3;
  float* sig_f = ws;               ws += NSAMP;

  coarse_depths_kernel<<<(NSAMP + 255) / 256, 256, 0, stream>>>(noise, dep_c);

  render_pass_kernel<<<512, 256, 0, stream>>>(planes, ray_o, ray_d, dep_c,
                                              W1, b1, W2, b2, rgb_c, sig_c);

  importance_kernel<<<(NRAYS + 127) / 128, 128, 0, stream>>>(dep_c, sig_c,
                                                             u_imp, dep_f);

  render_pass_kernel<<<512, 256, 0, stream>>>(planes, ray_o, ray_d, dep_f,
                                              W1, b1, W2, b2, rgb_f, sig_f);

  composite_kernel<<<(NRAYS + 127) / 128, 128, 0, stream>>>(
      dep_c, rgb_c, sig_c, dep_f, rgb_f, sig_f, out);
}
